// ClassificationNCA_75703093559477
// MI455X (gfx1250) — matrix operand order repaired and checked by reading, not compiled, not run on hardware
//
#include <hip/hip_runtime.h>
#include <cstdint>
#include <cstddef>

// ---------------------------------------------------------------------------
// ClassificationNCA on gfx1250: WMMA f16 (16x16x32) batched-MLP formulation.
// M = 16 pixels per wave, K/N tiled per layer (88->96 padded, 256, 128, 29->32).
// ---------------------------------------------------------------------------

#define NCH     29
#define N_IMG   3
#define N_HID   16
#define HDIM    128
#define IMGS    16
#define HH      64
#define WW      64
#define INVEC   88            // 29*3 + 1
#define KPAD1   96            // 3 chunks of 32
#define STEPS   20
#define TILES   (IMGS * HH * (WW / 16))   // 4096 wave-tiles of 16 pixels

typedef __attribute__((ext_vector_type(16))) _Float16 v16h;
typedef __attribute__((ext_vector_type(8)))  _Float16 v8h;
typedef __attribute__((ext_vector_type(8)))  float    v8f;

union V16U { v16h v; v8h h[2]; };

__device__ __forceinline__ v8f wmma_f16(v16h a, v16h b, v8f c) {
  // D = A(16x32 f16) x B(32x16 f16) + C(16x16 f32)
  return __builtin_amdgcn_wmma_f32_16x16x32_f16(
      /*neg_a=*/false, a, /*neg_b=*/false, b,
      /*c_mod=*/(short)0, c, /*reuse_a=*/false, /*reuse_b=*/false);
}

// A-fragment k offset for 16-bit A (ISA 7.12.2): lane<16: e0..7->K0..7, e8..15->K16..23
// lane>=16 shifts by 8.
__device__ __forceinline__ int koffA(int lane, int e) {
  return 8 * (lane >> 4) + (e < 8 ? e : e + 8);
}
// B-fragment k offset (lane holds column n=lane&15; same k order as koffA)
__device__ __forceinline__ int koffB(int lane, int e) {
  return 8 * (lane >> 4) + (e < 8 ? e : e + 8);
}

__device__ __forceinline__ uint32_t hash_u32(uint32_t v) {
  v ^= v >> 16; v *= 0x7feb352du;
  v ^= v >> 15; v *= 0x846ca68bu;
  v ^= v >> 16;
  return v;
}
__device__ __forceinline__ float hash_unif(uint32_t v) {
  return (float)(hash_u32(v) >> 8) * (1.0f / 16777216.0f);
}

__device__ __forceinline__ float state_at(const float* s, int b, int c, int y, int x) {
  if ((unsigned)y >= HH || (unsigned)x >= WW) return 0.0f;
  return s[(((size_t)b * NCH + c) * HH + y) * WW + x];
}

// sobel_x = outer([1,2,1],[-1,0,1])/8 ; sobel_y = transpose. XLA conv = cross-corr.
__device__ float sobel(const float* s, int b, int c, int y, int x, int isY) {
  const float rowv[3] = {1.0f, 2.0f, 1.0f};
  const float colv[3] = {-1.0f, 0.0f, 1.0f};
  float acc = 0.0f;
  #pragma unroll
  for (int r = 0; r < 3; ++r) {
    #pragma unroll
    for (int cc = 0; cc < 3; ++cc) {
      float w = isY ? (rowv[cc] * colv[r]) : (rowv[r] * colv[cc]);
      if (w != 0.0f) acc += w * state_at(s, b, c, y + r - 1, x + cc - 1);
    }
  }
  return acc * 0.125f;
}

// ---------------------------------------------------------------------------
// Kernel 1: state init. c<3: image; 3..18: ~N(0.5, 0.225^2) via hash; else 0.
// ---------------------------------------------------------------------------
__global__ __launch_bounds__(256) void nca_init(const float* __restrict__ xin,
                                                float* __restrict__ st) {
  int id = blockIdx.x * 256 + threadIdx.x;
  const int total = IMGS * NCH * HH * WW;
  if (id >= total) return;
  int x = id & (WW - 1);
  int y = (id >> 6) & (HH - 1);
  int c = (id >> 12) % NCH;
  int b = id / (NCH * HH * WW);
  float v;
  if (c < N_IMG) {
    v = xin[(((size_t)b * N_IMG + c) * HH + y) * WW + x];
  } else if (c < N_IMG + N_HID) {
    uint32_t s0 = (uint32_t)id * 4u + 0x9e3779b9u;
    float g = (hash_unif(s0) + hash_unif(s0 + 1) + hash_unif(s0 + 2) +
               hash_unif(s0 + 3) - 2.0f) * 1.7320508f;   // Irwin-Hall ~N(0,1)
    v = 0.5f + 0.225f * g;
  } else {
    v = 0.0f;
  }
  st[id] = v;
}

// ---------------------------------------------------------------------------
// Kernel 2: weight prep -> f16 WMMA B-fragments in global ws (L2 resident).
// w1: 16 ntiles x 3 kchunks ; w2: 8 x 8 ; w3: 2 x 4. 512 halves per fragment.
// ---------------------------------------------------------------------------
__global__ __launch_bounds__(256) void nca_prep(const float* __restrict__ w1,
                                                const float* __restrict__ w2,
                                                const float* __restrict__ w3,
                                                _Float16* __restrict__ w1f,
                                                _Float16* __restrict__ w2f,
                                                _Float16* __restrict__ w3f) {
  int id = blockIdx.x * 256 + threadIdx.x;
  const int SZ1 = 16 * 3 * 512, SZ2 = 8 * 8 * 512, SZ3 = 2 * 4 * 512;
  if (id < SZ1) {
    int e = id & 15, l = (id >> 4) & 31, q = (id >> 9) % 3, nt = (id >> 9) / 3;
    int n = nt * 16 + (l & 15);
    int k = q * 32 + koffB(l, e);
    w1f[id] = (_Float16)((k < INVEC) ? w1[(size_t)n * INVEC + k] : 0.0f);
  } else if (id < SZ1 + SZ2) {
    int j = id - SZ1;
    int e = j & 15, l = (j >> 4) & 31, q = (j >> 9) & 7, nt = j >> 12;
    int n = nt * 16 + (l & 15);
    int k = q * 32 + koffB(l, e);
    w2f[j] = (_Float16)w2[(size_t)n * (2 * HDIM) + k];
  } else if (id < SZ1 + SZ2 + SZ3) {
    int j = id - SZ1 - SZ2;
    int e = j & 15, l = (j >> 4) & 31, q = (j >> 9) & 3, nt = j >> 11;
    int n = nt * 16 + (l & 15);
    int k = q * 32 + koffB(l, e);
    w3f[j] = (_Float16)((n < NCH) ? w3[(size_t)n * HDIM + k] : 0.0f);
  }
}

// ---------------------------------------------------------------------------
// Kernel 3: perception -> f16 A-fragments (tile-major, lane-contiguous).
// One thread per f16 element. perc[((T*3+q)*32+l)*16+e].
// ---------------------------------------------------------------------------
__global__ __launch_bounds__(256) void nca_percept(const float* __restrict__ st,
                                                   _Float16* __restrict__ perc,
                                                   int step) {
  int id = blockIdx.x * 256 + threadIdx.x;
  if (id >= TILES * 3 * 512) return;
  int e = id & 15;
  int l = (id >> 4) & 31;
  int q = ((id >> 9)) % 3;
  int T = id / (3 * 512);
  int m = l & 15;
  int k = q * 32 + koffA(l, e);
  int b = T >> 8;
  int y = (T >> 2) & (HH - 1);
  int x = ((T & 3) << 4) + m;
  float f;
  if (k >= INVEC) {
    f = 0.0f;
  } else if (k == INVEC - 1) {
    f = (float)step * 0.01f;
  } else if (k < NCH) {
    f = state_at(st, b, k, y, x);
  } else if (k < 2 * NCH) {
    f = sobel(st, b, k - NCH, y, x, /*isY=*/0);
  } else {
    f = sobel(st, b, k - 2 * NCH, y, x, /*isY=*/1);
  }
  perc[id] = (_Float16)f;
}

// ---------------------------------------------------------------------------
// Kernel 4: MLP step. 8 waves/block, 16 pixels/wave. 120 WMMAs per wave.
// Dynamic LDS: 8 waves x 16 x 256 halves = 64 KB (C->A fragment transpose).
// ---------------------------------------------------------------------------
__global__ __launch_bounds__(256) void nca_mlp_step(const float* __restrict__ sin,
                                                    float* __restrict__ sout,
                                                    const _Float16* __restrict__ perc,
                                                    const _Float16* __restrict__ w1f,
                                                    const _Float16* __restrict__ w2f,
                                                    const _Float16* __restrict__ w3f,
                                                    const float* __restrict__ b1,
                                                    int step) {
  extern __shared__ _Float16 hb[];          // [8][16][256]
  const int lane = threadIdx.x & 31;
  const int wv   = threadIdx.x >> 5;
  const int T    = blockIdx.x * 8 + wv;
  const int b    = T >> 8;
  const int y    = (T >> 2) & (HH - 1);
  const int x0   = (T & 3) << 4;
  const int nrow = lane & 15;
  const int hi   = lane >> 4;
  _Float16* myh = &hb[(size_t)wv * 16 * 256];

  const v16h* pv  = (const v16h*)perc;
  const v16h* w1v = (const v16h*)w1f;
  const v16h* w2v = (const v16h*)w2f;
  const v16h* w3v = (const v16h*)w3f;

  // ---- layer 1: perc(16x96) x w1f -> h1(16x256), bias + leaky ----
  v16h aP[3];
  #pragma unroll
  for (int q = 0; q < 3; ++q) aP[q] = pv[((size_t)T * 3 + q) * 32 + lane];

  #pragma unroll
  for (int nt = 0; nt < 16; ++nt) {
    float bias = b1[nt * 16 + nrow];
    v8f c;
    #pragma unroll
    for (int i = 0; i < 8; ++i) c[i] = bias;
    #pragma unroll
    for (int q = 0; q < 3; ++q)
      c = wmma_f16(aP[q], w1v[((size_t)nt * 3 + q) * 32 + lane], c);
    #pragma unroll
    for (int i = 0; i < 8; ++i) c[i] = (c[i] >= 0.0f) ? c[i] : 0.01f * c[i];
    #pragma unroll
    for (int v = 0; v < 8; ++v)
      myh[(size_t)(v + 8 * hi) * 256 + nt * 16 + nrow] = (_Float16)c[v];
  }
  __syncthreads();

  // ---- layer 2: h1(16x256) x w2f -> h2(16x128), leaky ----
  v16h A2[8];
  #pragma unroll
  for (int q = 0; q < 8; ++q) {
    V16U u;
    u.h[0] = *(const v8h*)&myh[(size_t)nrow * 256 + q * 32 + 8 * hi];
    u.h[1] = *(const v8h*)&myh[(size_t)nrow * 256 + q * 32 + 16 + 8 * hi];
    A2[q] = u.v;
  }
  #pragma unroll
  for (int nt = 0; nt < 8; ++nt) {
    v8f c;
    #pragma unroll
    for (int i = 0; i < 8; ++i) c[i] = 0.0f;
    #pragma unroll
    for (int q = 0; q < 8; ++q)
      c = wmma_f16(A2[q], w2v[((size_t)nt * 8 + q) * 32 + lane], c);
    #pragma unroll
    for (int i = 0; i < 8; ++i) c[i] = (c[i] >= 0.0f) ? c[i] : 0.01f * c[i];
    #pragma unroll
    for (int v = 0; v < 8; ++v)
      myh[(size_t)(v + 8 * hi) * 256 + nt * 16 + nrow] = (_Float16)c[v];
  }
  __syncthreads();

  // ---- layer 3: h2(16x128) x w3f -> dx(16x32), fire-mask residual ----
  v16h A3[4];
  #pragma unroll
  for (int q = 0; q < 4; ++q) {
    V16U u;
    u.h[0] = *(const v8h*)&myh[(size_t)nrow * 256 + q * 32 + 8 * hi];
    u.h[1] = *(const v8h*)&myh[(size_t)nrow * 256 + q * 32 + 16 + 8 * hi];
    A3[q] = u.v;
  }
  #pragma unroll
  for (int nt = 0; nt < 2; ++nt) {
    v8f d;
    #pragma unroll
    for (int i = 0; i < 8; ++i) d[i] = 0.0f;
    #pragma unroll
    for (int q = 0; q < 4; ++q)
      d = wmma_f16(A3[q], w3v[((size_t)nt * 4 + q) * 32 + lane], d);

    int n = nt * 16 + nrow;                 // output channel
    if (n < NCH) {
      #pragma unroll
      for (int v = 0; v < 8; ++v) {
        int m = v + 8 * hi;
        int x = x0 + m;
        size_t si = (((size_t)b * NCH + n) * HH + y) * WW + x;
        float val = sin[si];
        if (n >= N_IMG) {                   // chan_mask
          uint32_t seed = ((((uint32_t)step * IMGS + b) * HH + y) * WW + x) ^ 0xF17Eu;
          if (hash_unif(seed) < 0.5f) val += d[v];   // fire mask
        }
        sout[si] = val;
      }
    }
  }
}

// ---------------------------------------------------------------------------
// Kernel 5: spatial mean of class channels + softmax. One block per image.
// ---------------------------------------------------------------------------
__global__ __launch_bounds__(256) void nca_reduce(const float* __restrict__ st,
                                                  float* __restrict__ out) {
  __shared__ float red[256];
  __shared__ float logits[10];
  int b = blockIdx.x, t = threadIdx.x;
  for (int c = 0; c < 10; ++c) {
    const float* p = st + (((size_t)b * NCH + (N_IMG + N_HID + c)) * HH * WW);
    float acc = 0.0f;
    for (int i = t; i < HH * WW; i += 256) acc += p[i];
    red[t] = acc;
    __syncthreads();
    for (int off = 128; off > 0; off >>= 1) {
      if (t < off) red[t] += red[t + off];
      __syncthreads();
    }
    if (t == 0) logits[c] = red[0] * (1.0f / (HH * WW));
    __syncthreads();
  }
  if (t == 0) {
    float mx = logits[0];
    for (int c = 1; c < 10; ++c) mx = fmaxf(mx, logits[c]);
    float se = 0.0f, ex[10];
    for (int c = 0; c < 10; ++c) { ex[c] = __expf(logits[c] - mx); se += ex[c]; }
    float inv = 1.0f / se;
    for (int c = 0; c < 10; ++c) out[b * 10 + c] = ex[c] * inv;
  }
}

// ---------------------------------------------------------------------------
extern "C" void kernel_launch(void* const* d_in, const int* in_sizes, int n_in,
                              void* d_out, int out_size, void* d_ws, size_t ws_size,
                              hipStream_t stream) {
  const float* x  = (const float*)d_in[0];
  const float* w1 = (const float*)d_in[1];
  const float* b1 = (const float*)d_in[2];
  const float* w2 = (const float*)d_in[3];
  const float* w3 = (const float*)d_in[4];
  (void)in_sizes; (void)n_in; (void)out_size; (void)ws_size;

  char* ws = (char*)d_ws;
  size_t off = 0;
  auto take = [&](size_t bytes) { size_t o = off; off = (off + bytes + 255) & ~(size_t)255; return o; };

  const size_t STATE_ELEMS = (size_t)IMGS * NCH * HH * WW;       // 1,900,544
  float*    sA   = (float*)(ws + take(STATE_ELEMS * sizeof(float)));
  float*    sB   = (float*)(ws + take(STATE_ELEMS * sizeof(float)));
  _Float16* perc = (_Float16*)(ws + take((size_t)TILES * 3 * 512 * sizeof(_Float16)));
  _Float16* w1f  = (_Float16*)(ws + take((size_t)16 * 3 * 512 * sizeof(_Float16)));
  _Float16* w2f  = (_Float16*)(ws + take((size_t)8 * 8 * 512 * sizeof(_Float16)));
  _Float16* w3f  = (_Float16*)(ws + take((size_t)2 * 4 * 512 * sizeof(_Float16)));

  nca_init<<<(int)(STATE_ELEMS / 256), 256, 0, stream>>>(x, sA);
  nca_prep<<<240, 256, 0, stream>>>(w1, w2, w3, w1f, w2f, w3f);

  float* cur = sA;
  float* nxt = sB;
  const int percThreads = TILES * 3 * 512;
  const size_t ldsBytes = (size_t)8 * 16 * 256 * sizeof(_Float16);   // 64 KB
  for (int step = 0; step < STEPS; ++step) {
    nca_percept<<<percThreads / 256, 256, 0, stream>>>(cur, perc, step);
    nca_mlp_step<<<TILES / 8, 256, ldsBytes, stream>>>(cur, nxt, perc, w1f, w2f, w3f, b1, step);
    float* t = cur; cur = nxt; nxt = t;
  }
  nca_reduce<<<IMGS, 256, 0, stream>>>(cur, (float*)d_out);
}
